// CostumRNN_Decoder_52252572123873
// MI455X (gfx1250) — matrix operand order repaired and checked by reading, not compiled, not run on hardware
//
#include <hip/hip_runtime.h>
#include <math.h>

// Problem constants (match reference)
constexpr int kB = 64;     // batch
constexpr int kH = 2;      // input channels
constexpr int kT = 2048;   // time
constexpr int kC = 256;    // hidden / output channels
constexpr float kEps = 1e-5f;

constexpr int kBT   = 4;            // batch tiles of 16 rows (WMMA M)
constexpr int kTile = 16 * kC;      // one (16 x 256) activation tile
constexpr int kWMatDw = 16 * 8 * 32 * 8; // frag dwords per 256x256 bf16 matrix = 32768
constexpr int kHalfMatDw = kWMatDw / 2;  // 8 ntiles worth = 16384 dwords (64 KB)

typedef __attribute__((ext_vector_type(8)))  float  v8f;
typedef __attribute__((ext_vector_type(16))) __bf16 v16bf;

// ---- float -> bf16 pair packed into one dword (RNE) ----
#if defined(__has_builtin) && __has_builtin(__builtin_amdgcn_cvt_pk_bf16_f32)
__device__ __forceinline__ unsigned bf16_2(float lo, float hi) {
  typedef __attribute__((ext_vector_type(2))) __bf16 v2bf;
  v2bf t = __builtin_amdgcn_cvt_pk_bf16_f32(lo, hi);
  unsigned r;
  __builtin_memcpy(&r, &t, 4);
  return r;
}
#else
__device__ __forceinline__ unsigned bf16_1(float f) {
  unsigned u = __float_as_uint(f);
  return (u + 0x7fffu + ((u >> 16) & 1u)) >> 16;
}
__device__ __forceinline__ unsigned bf16_2(float lo, float hi) {
  return bf16_1(lo) | (bf16_1(hi) << 16);
}
#endif

__device__ __forceinline__ float sigmoid_fast(float v) {
  return 1.0f / (1.0f + __expf(-v));
}
__device__ __forceinline__ float tanh_fast(float v) {
  float c = fminf(fmaxf(v, -15.f), 15.f);
  float e = __expf(2.0f * c);
  return (e - 1.0f) / (e + 1.0f);
}

// =====================================================================
// Kernel 1: per-batch moments of x over T
// =====================================================================
__global__ __launch_bounds__(256) void stats_kernel(const float* __restrict__ x,
                                                    float* __restrict__ stats) {
  const int b = blockIdx.x;
  const int tid = threadIdx.x;
  const float* x0 = x + (size_t)(b * kH + 0) * kT;
  const float* x1 = x + (size_t)(b * kH + 1) * kT;
  float s0 = 0.f, s1 = 0.f, q0 = 0.f, q1 = 0.f, pp = 0.f;
  for (int t = tid; t < kT; t += 256) {
    float a = x0[t], c = x1[t];
    s0 += a; s1 += c; q0 += a * a; q1 += c * c; pp += a * c;
  }
  __shared__ float red[5][256];
  red[0][tid] = s0; red[1][tid] = s1; red[2][tid] = q0; red[3][tid] = q1; red[4][tid] = pp;
  __syncthreads();
  for (int off = 128; off > 0; off >>= 1) {
    if (tid < off) {
      #pragma unroll
      for (int r = 0; r < 5; ++r) red[r][tid] += red[r][tid + off];
    }
    __syncthreads();
  }
  if (tid == 0) {
    const float inv = 1.0f / (float)kT;
    #pragma unroll
    for (int r = 0; r < 5; ++r) stats[b * 8 + r] = red[r][0] * inv;
  }
}

// =====================================================================
// Kernel 2: xu = ReLU(InstanceNorm(conv1x1(x))) in recurrence tile order
// =====================================================================
__global__ __launch_bounds__(256) void xu_kernel(
    const float* __restrict__ x, const float* __restrict__ conv_w,
    const float* __restrict__ conv_b, const float* __restrict__ gamma,
    const float* __restrict__ beta, const float* __restrict__ stats,
    float* __restrict__ xu) {
  const int blk = blockIdx.x;          // bt*T + t
  const int bt  = blk >> 11;
  const int t   = blk & (kT - 1);
  const int c   = threadIdx.x;
  const float w0 = conv_w[c * kH + 0];
  const float w1 = conv_w[c * kH + 1];
  const float bc = conv_b[c];
  const float g  = gamma[c];
  const float be = beta[c];
  float* dst = xu + (size_t)blk * kTile;
  #pragma unroll 4
  for (int m = 0; m < 16; ++m) {
    const int b = bt * 16 + m;
    const float* st = stats + b * 8;
    const float mu  = w0 * st[0] + w1 * st[1] + bc;
    const float exu2 = w0 * w0 * st[2] + w1 * w1 * st[3] + 2.f * w0 * w1 * st[4]
                     + 2.f * bc * (w0 * st[0] + w1 * st[1]) + bc * bc;
    const float var = exu2 - mu * mu;
    const float xv0 = x[(size_t)(b * kH + 0) * kT + t];
    const float xv1 = x[(size_t)(b * kH + 1) * kT + t];
    const float val = w0 * xv0 + w1 * xv1 + bc;
    float o = (val - mu) * rsqrtf(var + kEps) * g + be;
    dst[m * kC + c] = fmaxf(o, 0.f);
  }
}

// =====================================================================
// Kernel 3: quantize weights to bf16, swizzle to WMMA B-fragment layout
// =====================================================================
__global__ __launch_bounds__(256) void wfrag_kernel(
    const float* __restrict__ Wr, const float* __restrict__ Wu,
    const float* __restrict__ Wp, const float* __restrict__ Wrec,
    unsigned* __restrict__ wf) {
  const int d = blockIdx.x * 256 + threadIdx.x;   // 0 .. 32767
  const int nt = d >> 11;
  const int kf = (d >> 8) & 7;
  const int ln = (d >> 3) & 31;
  const int j  = d & 7;
  const int n  = nt * 16 + (ln & 15);
  const int k  = kf * 32 + ((ln >> 4) << 3) + ((j >> 2) << 4) + ((j & 3) << 1);
  const float* Ws[4] = {Wr, Wu, Wp, Wrec};
  #pragma unroll
  for (int mtx = 0; mtx < 4; ++mtx) {
    const float* W = Ws[mtx];
    wf[mtx * kWMatDw + d] = bf16_2(W[n * kC + k], W[n * kC + k + 1]);
  }
}

// ---- pack 16x256 fp32 row-major LDS tile into WMMA A-fragments -------
__device__ __forceinline__ void pack1_frag(unsigned* dst, const float* A, int tid) {
  const int pm  = tid >> 4;
  const int pk0 = (tid & 15) << 4;
  const int kf  = pk0 >> 5;
  const int jb  = ((pk0 >> 4) & 1) << 2;   // 0 or 4
  const float2* a2 = (const float2*)A;
  const int base = (pm * kC + pk0) >> 1;
  uint4 w0, w1;
  unsigned* p0 = (unsigned*)&w0;
  unsigned* p1 = (unsigned*)&w1;
  #pragma unroll
  for (int d = 0; d < 4; ++d) {
    float2 lo = a2[base + d];
    float2 hi = a2[base + 4 + d];
    p0[d] = bf16_2(lo.x, lo.y);
    p1[d] = bf16_2(hi.x, hi.y);
  }
  *(uint4*)&dst[(kf << 8) + (pm << 3) + jb]        = w0;
  *(uint4*)&dst[(kf << 8) + ((pm + 16) << 3) + jb] = w1;
}
__device__ __forceinline__ void pack2_frag(unsigned* dst, const float* A,
                                           const float* B, int tid) {
  const int pm  = tid >> 4;
  const int pk0 = (tid & 15) << 4;
  const int kf  = pk0 >> 5;
  const int jb  = ((pk0 >> 4) & 1) << 2;
  const float2* a2 = (const float2*)A;
  const float2* b2 = (const float2*)B;
  const int base = (pm * kC + pk0) >> 1;
  uint4 w0, w1;
  unsigned* p0 = (unsigned*)&w0;
  unsigned* p1 = (unsigned*)&w1;
  #pragma unroll
  for (int d = 0; d < 4; ++d) {
    float2 la = a2[base + d],     lb = b2[base + d];
    float2 ha = a2[base + 4 + d], hb = b2[base + 4 + d];
    p0[d] = bf16_2(la.x + lb.x, la.y + lb.y);
    p1[d] = bf16_2(ha.x + hb.x, ha.y + hb.y);
  }
  *(uint4*)&dst[(kf << 8) + (pm << 3) + jb]        = w0;
  *(uint4*)&dst[(kf << 8) + ((pm + 16) << 3) + jb] = w1;
}

#define WMMA_BF16(A, Bv, Cv) \
  __builtin_amdgcn_wmma_f32_16x16x32_bf16(false, (A), false, (Bv), (short)0, (Cv), false, false)

// =====================================================================
// Kernel 4: the recurrence.
//  - Register-resident weights: Wr (all), Wu (even ntiles), Wp (all) = 320 VGPRs
//  - LDS weights: Wrec (128 KB) + Wu odd ntiles (64 KB)
//  - A-fragments software-pipelined (double buffered) in each GEMM phase
//  - 4 barriers/step, x tile prefetched via global_prefetch
// =====================================================================
__global__ __launch_bounds__(256, 1) void rnn_kernel(
    const float* __restrict__ xu, const unsigned* __restrict__ wf,
    const float* __restrict__ br, const float* __restrict__ bu,
    const float* __restrict__ bp, const float* __restrict__ brec,
    const float* __restrict__ h0, float* __restrict__ out) {
  __shared__ alignas(32) float    xt[kTile];        // current x_t tile
  __shared__ alignas(32) float    hbuf[kTile];      // hidden state
  __shared__ alignas(32) float    arow[kTile];      // x_t + r*h staging
  __shared__ alignas(32) unsigned fs[8 * 256];      // A-frags of s = x+h
  __shared__ alignas(32) unsigned fa[8 * 256];      // A-frags of x + r*h
  __shared__ alignas(32) unsigned fh[8 * 256];      // A-frags of h_new
  __shared__ alignas(32) unsigned wzl[kWMatDw];     // Wrec B-frags (128 KB)
  __shared__ alignas(32) unsigned wul[kHalfMatDw];  // Wu odd-ntile B-frags (64 KB)

  const int tid  = threadIdx.x;
  const int lane = tid & 31;
  const int wv   = tid >> 5;     // wave 0..7
  const int bt   = blockIdx.x;   // 0..3

  const int mBase = (lane >> 4) << 3;
  const int nSub  = lane & 15;

  float biasr[2], biasu[2], biasp[2], biasz[2];
  #pragma unroll
  for (int q = 0; q < 2; ++q) {
    const int n = ((wv * 2 + q) << 4) + nSub;
    biasr[q] = br[n]; biasu[q] = bu[n]; biasp[q] = bp[n]; biasz[q] = brec[n];
  }

  const unsigned* wfR = wf + 0 * kWMatDw;
  const unsigned* wfU = wf + 1 * kWMatDw;
  const unsigned* wfP = wf + 2 * kWMatDw;

  // ---- prologue: Wrec -> LDS, Wu odd ntiles -> LDS, h init, x_{T-1} ----
  {
    const uint4* s = (const uint4*)(wf + 3 * kWMatDw);
    uint4* d = (uint4*)wzl;
    for (int i = tid; i < kWMatDw / 4; i += 256) d[i] = s[i];
  }
  {
    const uint4* s = (const uint4*)wfU;
    uint4* d = (uint4*)wul;
    for (int i = tid; i < kHalfMatDw / 4; i += 256) {
      const int nt2 = i >> 9;        // 512 uint4 per ntile
      const int rem = i & 511;
      d[i] = s[((nt2 * 2 + 1) << 9) + rem];   // odd ntiles of Wu
    }
  }
  for (int e = tid; e < kTile; e += 256) hbuf[e] = h0[e & (kC - 1)];
  {
    const float4* s = (const float4*)(xu + (size_t)(bt * kT + (kT - 1)) * kTile);
    float4* xt4 = (float4*)xt;
    #pragma unroll
    for (int i = 0; i < 4; ++i) xt4[tid + i * 256] = s[tid + i * 256];
  }
  __syncthreads();
  pack2_frag(fs, xt, hbuf, tid);      // s = x_{T-1} + h0
  __syncthreads();

  for (int tp = 0; tp < kT; ++tp) {
    const int t = kT - 1 - tp;        // actual time index being processed
    const int tn = (tp + 1 < kT) ? (kT - 2 - tp) : 0;   // next tile index
    const float* nsrc = xu + (size_t)(bt * kT + tn) * kTile;

    // prefetch next x tile into caches (zero register cost)
    __builtin_prefetch((const char*)nsrc + tid * 64, 0, 1);

    // ---- P1: r and u GEMMs (A = fs, double-buffered) ----
    v8f accR[2], accU[2];
    #pragma unroll
    for (int q = 0; q < 2; ++q) {
      const int nt = wv * 2 + q;
      v8f cr = {}; v8f cu = {};
      const unsigned* pr = wfR + (nt << 11);
      const unsigned* pu = (q == 0) ? (wfU + (nt << 11))
                                    : (const unsigned*)(wul + (wv << 11));
      v16bf a = *(const v16bf*)&fs[lane << 3];
      #pragma unroll
      for (int kf = 0; kf < 8; ++kf) {
        v16bf an = a;
        if (kf < 7) an = *(const v16bf*)&fs[((kf + 1) << 8) + (lane << 3)];
        v16bf b0 = *(const v16bf*)&pr[(kf << 8) + (lane << 3)];
        v16bf b1 = *(const v16bf*)&pu[(kf << 8) + (lane << 3)];
        cr = WMMA_BF16(a, b0, cr);
        cu = WMMA_BF16(a, b1, cu);
        a = an;
      }
      accR[q] = cr; accU[q] = cu;
    }

    // ---- P2: gates; stage arow = x_t + r*h ----
    #pragma unroll
    for (int q = 0; q < 2; ++q) {
      #pragma unroll
      for (int v = 0; v < 8; ++v) {
        const int m = mBase + v;
        const int n = ((wv * 2 + q) << 4) + nSub;
        const int idx = m * kC + n;
        const float r = sigmoid_fast(accR[q][v] + biasr[q]);
        accU[q][v] = sigmoid_fast(accU[q][v] + biasu[q]);
        arow[idx] = xt[idx] + r * hbuf[idx];
      }
    }
    __syncthreads();                                   // B1
    pack1_frag(fa, arow, tid);
    __syncthreads();                                   // B2

    // ---- P4: h_hat GEMM (double-buffered A), gate combine, h update ----
    #pragma unroll
    for (int q = 0; q < 2; ++q) {
      const int nt = wv * 2 + q;
      v8f cp = {};
      const unsigned* pp = wfP + (nt << 11);
      v16bf a = *(const v16bf*)&fa[lane << 3];
      #pragma unroll
      for (int kf = 0; kf < 8; ++kf) {
        v16bf an = a;
        if (kf < 7) an = *(const v16bf*)&fa[((kf + 1) << 8) + (lane << 3)];
        v16bf b = *(const v16bf*)&pp[(kf << 8) + (lane << 3)];
        cp = WMMA_BF16(a, b, cp);
        a = an;
      }
      #pragma unroll
      for (int v = 0; v < 8; ++v) {
        const int m = mBase + v;
        const int n = (nt << 4) + nSub;
        const int idx = m * kC + n;
        const float hh = tanh_fast(cp[v] + biasp[q]);
        const float u  = accU[q][v];
        hbuf[idx] = (1.0f - u) * hbuf[idx] + u * hh;
      }
    }
    {  // copy next x tile (prefetched) into xt
      const float4* s = (const float4*)nsrc;
      float4* xt4 = (float4*)xt;
      #pragma unroll
      for (int i = 0; i < 4; ++i) xt4[tid + i * 256] = s[tid + i * 256];
    }
    __syncthreads();                                   // B3

    // ---- P5: pack h_new (for z) and s_next = x_next + h_new ----
    pack1_frag(fh, hbuf, tid);
    pack2_frag(fs, xt, hbuf, tid);
    __syncthreads();                                   // B4

    // ---- P6: z GEMM (A+B from LDS, double-buffered A) + store ----
    #pragma unroll
    for (int q = 0; q < 2; ++q) {
      const int nt = wv * 2 + q;
      v8f cz = {};
      const unsigned* pz = wzl + (nt << 11);
      v16bf a = *(const v16bf*)&fh[lane << 3];
      #pragma unroll
      for (int kf = 0; kf < 8; ++kf) {
        v16bf an = a;
        if (kf < 7) an = *(const v16bf*)&fh[((kf + 1) << 8) + (lane << 3)];
        v16bf b = *(const v16bf*)&pz[(kf << 8) + (lane << 3)];
        cz = WMMA_BF16(a, b, cz);
        a = an;
      }
      #pragma unroll
      for (int v = 0; v < 8; ++v) {
        const int m = mBase + v;
        const int n = (nt << 4) + nSub;
        const size_t o = (size_t)(bt * 16 + m) * (kC * kT) + (size_t)n * kT + t;
        out[o] = cz[v] + biasz[q];
      }
    }
  }
}

// =====================================================================
extern "C" void kernel_launch(void* const* d_in, const int* in_sizes, int n_in,
                              void* d_out, int out_size, void* d_ws, size_t ws_size,
                              hipStream_t stream) {
  const float* x      = (const float*)d_in[0];
  const float* conv_w = (const float*)d_in[1];
  const float* conv_b = (const float*)d_in[2];
  const float* gamma  = (const float*)d_in[3];
  const float* beta   = (const float*)d_in[4];
  const float* Wr     = (const float*)d_in[5];
  const float* brv    = (const float*)d_in[6];
  const float* Wu     = (const float*)d_in[7];
  const float* buv    = (const float*)d_in[8];
  const float* Wp     = (const float*)d_in[9];
  const float* bpv    = (const float*)d_in[10];
  const float* Wrec   = (const float*)d_in[11];
  const float* brec   = (const float*)d_in[12];
  const float* h0     = (const float*)d_in[13];
  float* out = (float*)d_out;

  // workspace: [xu fp32: 128 MB][weight frags bf16: 512 KB][stats: 2 KB]
  float*    xu  = (float*)d_ws;
  unsigned* wfr = (unsigned*)(xu + (size_t)kBT * kT * kTile);
  float*    st  = (float*)(wfr + 4 * kWMatDw);

  stats_kernel<<<kB, 256, 0, stream>>>(x, st);
  xu_kernel<<<kBT * kT, 256, 0, stream>>>(x, conv_w, conv_b, gamma, beta, st, xu);
  wfrag_kernel<<<kWMatDw / 256, 256, 0, stream>>>(Wr, Wu, Wp, Wrec, wfr);
  rnn_kernel<<<kBT, 256, 0, stream>>>(xu, wfr, brv, buv, bpv, brec, h0, out);
}
